// NPUGroupGemmOp_9526237463096
// MI455X (gfx1250) — matrix operand order repaired and checked by reading, not compiled, not run on hardware
//
#include <hip/hip_runtime.h>

typedef __attribute__((ext_vector_type(16))) __bf16 v16bf;
typedef __attribute__((ext_vector_type(8)))  __bf16 v8bf;
typedef __attribute__((ext_vector_type(8)))  float  v8f;

#define NEXP 32
#define MDIM 4096
#define KDIM 1024
#define NDIM 128

#define BM   128          // block tile M
#define BK   32           // K chunk == bf16 WMMA K
#define RECW 20           // u32s per LDS record: 32 bf16 (16 u32) + 4 u32 pad = 80B
                          // 80B stride => lane*80 mod 256 covers all 16 16B slots: conflict-free

// Split a fp32 pair into packed-bf16 hi (truncated; residual captured exactly by lo)
// and lo (round-half-away). v_perm_b32 packs the two high-16s in one instruction.
__device__ __forceinline__ void split2(float x0, float x1, unsigned& hi, unsigned& lo) {
    unsigned u0 = __float_as_uint(x0), u1 = __float_as_uint(x1);
    hi = __builtin_amdgcn_perm(u1, u0, 0x07060302u);   // {u1[31:16], u0[31:16]}
    float h0 = __uint_as_float(u0 & 0xFFFF0000u);
    float h1 = __uint_as_float(u1 & 0xFFFF0000u);
    unsigned l0 = __float_as_uint(x0 - h0) + 0x8000u;  // round-half-away to bf16
    unsigned l1 = __float_as_uint(x1 - h1) + 0x8000u;
    lo = __builtin_amdgcn_perm(l1, l0, 0x07060302u);
}

__device__ __forceinline__ v16bf frag2(const unsigned* p0, const unsigned* p1) {
    union { v8bf h[2]; v16bf v; } U;
    U.h[0] = *(const v8bf*)p0;
    U.h[1] = *(const v8bf*)p1;
    return U.v;
}

__global__ __launch_bounds__(256)
void grouped_gemm_splitbf16_wmma(const float* __restrict__ A,
                                 const float* __restrict__ B,
                                 float* __restrict__ C)
{
    // bf16 tiles in LDS, natural-K-order records of 32 bf16 (64B) + 16B pad
    __shared__ unsigned sAhi[BM   * RECW];   // A tile [row][k]
    __shared__ unsigned sAlo[BM   * RECW];
    __shared__ unsigned sBhi[NDIM * RECW];   // B tile transposed [n][k]
    __shared__ unsigned sBlo[NDIM * RECW];

    const int tid = threadIdx.x;
    const int e   = blockIdx.y;
    const int m0  = blockIdx.x * BM;

    const float* Ae = A + (size_t)e * MDIM * KDIM;
    const float* Be = B + (size_t)e * KDIM * NDIM;
    float*       Ce = C + (size_t)e * MDIM * NDIM;

    const int lane = tid & 31;           // wave32
    const int w    = tid >> 5;           // 8 waves
    const int half = lane >> 4;          // lane-half selects K interleave
    const int lrow = lane & 15;
    const int ms0  = (w & 3) * 2;        // 2 m-subtiles (16 rows) per wave
    const int ns0  = (w >> 2) * 4;       // 4 n-subtiles per wave

    // Hoisted per-thread load bases: i-loop offsets fold into immediates.
    const int aRow = tid >> 3;                    // 0..31 (+ i*32)
    const int aC4  = (tid & 7) * 4;               // k offset within chunk
    const float* pA = Ae + (size_t)(m0 + aRow) * KDIM + aC4;
    const int bN   = tid & 127;                   // column
    const int bK0  = (tid >> 7) * 4;              // 0 or 4 (+ i*8)
    const float* pB = Be + (size_t)bK0 * NDIM + bN;

    v8f acc[2][4] = {};

    const int KT = KDIM / BK;            // 32 K-chunks
    for (int kc = 0; kc < KT; ++kc) {
        const int kb = kc * BK;

        // ---- A tile 128x32: fp32 global (float4, coalesced) -> bf16 hi/lo LDS ----
        #pragma unroll
        for (int i = 0; i < 4; ++i) {
            float4 v = *(const float4*)(pA + kb + (size_t)i * 32 * KDIM);
            unsigned h0, l0, h1, l1;
            split2(v.x, v.y, h0, l0);
            split2(v.z, v.w, h1, l1);
            const int row = i * 32 + aRow;
            *(uint2*)&sAhi[row * RECW + (aC4 >> 1)] = make_uint2(h0, h1);
            *(uint2*)&sAlo[row * RECW + (aC4 >> 1)] = make_uint2(l0, l1);
        }
        // ---- B tile 32x128: 4 coalesced rows per thread, packed along K, transposed ----
        #pragma unroll
        for (int i = 0; i < 4; ++i) {
            const float* bp = pB + (size_t)(kb + i * 8) * NDIM;
            float x0 = bp[0 * NDIM], x1 = bp[1 * NDIM], x2 = bp[2 * NDIM], x3 = bp[3 * NDIM];
            unsigned h0, l0, h1, l1;
            split2(x0, x1, h0, l0);
            split2(x2, x3, h1, l1);
            const int kHalf = (bK0 >> 1) + i * 4;     // u32 index of k start
            *(uint2*)&sBhi[bN * RECW + kHalf] = make_uint2(h0, h1);
            *(uint2*)&sBlo[bN * RECW + kHalf] = make_uint2(l0, l1);
        }
        __syncthreads();

        // Prefetch next K tile while this chunk computes (global_prefetch_b8)
        if (kc + 1 < KT) {
            __builtin_prefetch(pA + kb + BK, 0, 3);
            __builtin_prefetch(pB + (size_t)(kb + BK) * NDIM, 0, 3);
        }

        // ---- fragments: pure ds_load_b128 pairs, no conversion VALU ----
        // A lane-half: K {h*8..h*8+7, 16+h*8..} -> record bytes half*16 and 32+half*16
        v16bf ahi[2], alo[2];
        #pragma unroll
        for (int mt = 0; mt < 2; ++mt) {
            int row = (ms0 + mt) * 16 + lrow;
            const unsigned* bh = &sAhi[row * RECW + half * 4];
            const unsigned* bl = &sAlo[row * RECW + half * 4];
            ahi[mt] = frag2(bh, bh + 8);
            alo[mt] = frag2(bl, bl + 8);
        }
        // B lane-half: K {h*8..h*8+7, 16+h*8..} -> record bytes half*16 and 32+half*16
        v16bf bhi[4], blo[4];
        #pragma unroll
        for (int nt = 0; nt < 4; ++nt) {
            int col = (ns0 + nt) * 16 + lrow;
            const unsigned* bh = &sBhi[col * RECW + half * 4];
            const unsigned* bl = &sBlo[col * RECW + half * 4];
            bhi[nt] = frag2(bh, bh + 8);
            blo[nt] = frag2(bl, bl + 8);
        }

        // ---- split-precision WMMA: C += Ahi*Bhi + Ahi*Blo + Alo*Bhi ----
        #pragma unroll
        for (int mt = 0; mt < 2; ++mt)
        #pragma unroll
        for (int nt = 0; nt < 4; ++nt) {
            acc[mt][nt] = __builtin_amdgcn_wmma_f32_16x16x32_bf16(
                false, ahi[mt], false, bhi[nt], (short)0, acc[mt][nt], false, false);
            acc[mt][nt] = __builtin_amdgcn_wmma_f32_16x16x32_bf16(
                false, ahi[mt], false, blo[nt], (short)0, acc[mt][nt], false, false);
            acc[mt][nt] = __builtin_amdgcn_wmma_f32_16x16x32_bf16(
                false, alo[mt], false, bhi[nt], (short)0, acc[mt][nt], false, false);
        }
        __syncthreads();
    }

    // ---- epilogue: C/D layout (M = v + 8*half within subtile) -> global ----
    #pragma unroll
    for (int mt = 0; mt < 2; ++mt)
    #pragma unroll
    for (int nt = 0; nt < 4; ++nt) {
        const int col = (ns0 + nt) * 16 + lrow;
        #pragma unroll
        for (int v = 0; v < 8; ++v) {
            const int row = m0 + (ms0 + mt) * 16 + half * 8 + v;
            Ce[(size_t)row * NDIM + col] = acc[mt][nt][v];
        }
    }
}

extern "C" void kernel_launch(void* const* d_in, const int* in_sizes, int n_in,
                              void* d_out, int out_size, void* d_ws, size_t ws_size,
                              hipStream_t stream) {
    const float* a = (const float*)d_in[0];
    const float* b = (const float*)d_in[1];
    float*       c = (float*)d_out;
    (void)in_sizes; (void)n_in; (void)out_size; (void)d_ws; (void)ws_size;

    dim3 grid(MDIM / BM, NEXP);   // 32 M-tiles x 32 experts
    dim3 block(256);              // 8 wave32s
    hipLaunchKernelGGL(grouped_gemm_splitbf16_wmma, grid, block, 0, stream, a, b, c);
}
